// MultiHeadAttention_90443421319193
// MI455X (gfx1250) — matrix operand order repaired and checked by reading, not compiled, not run on hardware
//
#include <hip/hip_runtime.h>

// ---------------------------------------------------------------------------
// MI455X / gfx1250 multi-head attention, bf16 WMMA (V_WMMA_F32_16X16X32_BF16)
// B=4, T=2048, H=16, D=64, C=1024.  All matmuls on the matrix pipe.
// Round 2: 32x64 GEMM tiles (2x A reuse), register-budget hints to kill the
// scratch spills seen in flash_attn's disassembly.
// ---------------------------------------------------------------------------

typedef __attribute__((ext_vector_type(16))) __bf16 v16bf;
typedef __attribute__((ext_vector_type(8)))  __bf16 v8bf;
typedef __attribute__((ext_vector_type(8)))  float  v8f;

#define BB   4
#define TT   2048
#define HH   16
#define DD   64
#define CC   1024
#define MM   (BB * TT)   // 8192

// ---- bf16 helpers (RNE convert) -------------------------------------------
__device__ __forceinline__ __bf16 f2bf(float f) {
  unsigned u = __float_as_uint(f);
  u += 0x7FFFu + ((u >> 16) & 1u);
  unsigned short s = (unsigned short)(u >> 16);
  return __builtin_bit_cast(__bf16, s);
}
__device__ __forceinline__ float bf2f(__bf16 h) {
  unsigned short s = __builtin_bit_cast(unsigned short, h);
  return __uint_as_float(((unsigned)s) << 16);
}
__device__ __forceinline__ v16bf cat16(v8bf a, v8bf b) {
  return __builtin_shufflevector(a, b, 0,1,2,3,4,5,6,7,8,9,10,11,12,13,14,15);
}
__device__ __forceinline__ v8bf ld8(const __bf16* p) {
  return *reinterpret_cast<const v8bf*>(p);
}
__device__ __forceinline__ v8f wmma_bf16(v16bf a, v16bf b, v8f c) {
  return __builtin_amdgcn_wmma_f32_16x16x32_bf16(false, a, false, b, (short)0, c,
                                                 false, false);
}

// ---- f32 -> bf16 bulk convert ---------------------------------------------
__global__ void cvt_f32_to_bf16(const float* __restrict__ in,
                                __bf16* __restrict__ out, int n) {
  int i = blockIdx.x * blockDim.x + threadIdx.x;
  if (i < n) out[i] = f2bf(in[i]);
}

// ---- W[K][N] f32 -> Wt[N][K] bf16 (LDS tile transpose) --------------------
__global__ void transpose_cvt(const float* __restrict__ W,
                              __bf16* __restrict__ Wt, int K, int N) {
  __shared__ float tile[32][33];
  int n0 = blockIdx.x * 32, k0 = blockIdx.y * 32;
  tile[threadIdx.y][threadIdx.x] =
      W[(size_t)(k0 + threadIdx.y) * N + n0 + threadIdx.x];
  __syncthreads();
  Wt[(size_t)(n0 + threadIdx.y) * K + k0 + threadIdx.x] =
      f2bf(tile[threadIdx.x][threadIdx.y]);
}

// ---- register-tiled WMMA GEMM: C[M,N] = A[M,K] @ Bt[N,K]^T (+bias) --------
// One wave per 32x64 tile: 8 f32 accumulators, each B-frag reused by 2 A-frags
// (~21 FLOP/B from cache).  256-VGPR budget (4 waves/SIMD) is plenty.
template <bool F32OUT>
__global__ void __launch_bounds__(256, 4)
wmma_gemm(const __bf16* __restrict__ A, const __bf16* __restrict__ Bt,
          const float* __restrict__ bias, void* __restrict__ Cout,
          int M, int N, int K) {
  const int lane = threadIdx.x & 31;
  const int wave = threadIdx.x >> 5;
  const int lh = lane & 15, hi = lane >> 4;
  const int m0 = (blockIdx.y * 8 + wave) * 32;
  const int n0 = blockIdx.x * 64;

  v8f acc0[4] = {}, acc1[4] = {};
  // A-frag: lane-half 0 holds row lh, K {k..k+7, k+16..k+23}; half 1: {+8,+24}
  const __bf16* Ap0 = A + (size_t)(m0 + lh) * K + hi * 8;
  const __bf16* Ap1 = A + (size_t)(m0 + 16 + lh) * K + hi * 8;
  const __bf16* Bp[4];
#pragma unroll
  for (int c = 0; c < 4; ++c)
    Bp[c] = Bt + (size_t)(n0 + c * 16 + lh) * K + hi * 8;

  for (int k = 0; k < K; k += 32) {
    if (k + 256 < K) {
      __builtin_prefetch(Ap0 + k + 256, 0, 1);
      __builtin_prefetch(Ap1 + k + 256, 0, 1);
    }
    v16bf af0 = cat16(ld8(Ap0 + k), ld8(Ap0 + k + 16));
    v16bf af1 = cat16(ld8(Ap1 + k), ld8(Ap1 + k + 16));
#pragma unroll
    for (int c = 0; c < 4; ++c) {
      v16bf bf = cat16(ld8(Bp[c] + k), ld8(Bp[c] + k + 16));
      acc0[c] = wmma_bf16(af0, bf, acc0[c]);
      acc1[c] = wmma_bf16(af1, bf, acc1[c]);
    }
  }
  // C layout: VGPR r -> row +r (lanes 0-15) / +r+8 (lanes 16-31), col = lane
#pragma unroll
  for (int r = 0; r < 8; ++r) {
    int row0 = m0 + r + hi * 8;
    int row1 = row0 + 16;
    size_t base0 = (size_t)row0 * N + n0 + lh;
    size_t base1 = (size_t)row1 * N + n0 + lh;
    if (F32OUT) {
      float* o = (float*)Cout;
#pragma unroll
      for (int c = 0; c < 4; ++c) {
        float bv = bias[n0 + c * 16 + lh];
        o[base0 + c * 16] = acc0[c][r] + bv;
        o[base1 + c * 16] = acc1[c][r] + bv;
      }
    } else {
      __bf16* o = (__bf16*)Cout;
#pragma unroll
      for (int c = 0; c < 4; ++c) {
        o[base0 + c * 16] = f2bf(acc0[c][r]);
        o[base1 + c * 16] = f2bf(acc1[c][r]);
      }
    }
  }
}

// ---- RoPE + layout pack ----------------------------------------------------
// qlin/klin/vlin: [B,T,H*D] bf16.  q,k -> [B,H,T,D] with RoPE.  v -> [B,H,D,T].
__global__ void rope_pack(const __bf16* __restrict__ qlin,
                          const __bf16* __restrict__ klin,
                          const __bf16* __restrict__ vlin,
                          __bf16* __restrict__ qr, __bf16* __restrict__ kr,
                          __bf16* __restrict__ vt) {
  int idx = blockIdx.x * blockDim.x + threadIdx.x;  // B*T*H*(D/2) threads
  int d2 = idx & 31;
  int h = (idx >> 5) & 15;
  int t = (idx >> 9) & (TT - 1);
  int b = idx >> 20;
  float inv = __powf(10000.0f, -((float)(2 * d2)) / (float)DD);
  float s, c;
  __sincosf((float)t * inv, &s, &c);
  size_t ib = ((size_t)(b * TT + t)) * CC + h * DD + 2 * d2;
  size_t ob = (((size_t)(b * HH + h)) * TT + t) * DD + 2 * d2;
  float q0 = bf2f(qlin[ib]), q1 = bf2f(qlin[ib + 1]);
  float k0 = bf2f(klin[ib]), k1 = bf2f(klin[ib + 1]);
  qr[ob] = f2bf(q0 * c - q1 * s);
  qr[ob + 1] = f2bf(q0 * s + q1 * c);
  kr[ob] = f2bf(k0 * c - k1 * s);
  kr[ob + 1] = f2bf(k0 * s + k1 * c);
  size_t vb = (((size_t)(b * HH + h)) * DD + 2 * d2) * TT + t;
  vt[vb] = vlin[ib];
  vt[vb + TT] = vlin[ib + 1];
}

// ---- causal flash attention, one wave per 16-query tile -------------------
// __launch_bounds__(256, 2): allow up to 512 VGPRs/wave so the Q fragments,
// 4 output accumulators and softmax state all stay resident (no scratch).
__global__ void __launch_bounds__(256, 2)
flash_attn(const __bf16* __restrict__ qr, const __bf16* __restrict__ kr,
           const __bf16* __restrict__ vt, __bf16* __restrict__ att) {
  __shared__ __align__(16) __bf16 pbuf[8 * 16 * 32];  // per-wave P tile
  const int lane = threadIdx.x & 31, wave = threadIdx.x >> 5;
  const int lh = lane & 15, hi = lane >> 4;
  const int bh = blockIdx.y, b = bh >> 4, h = bh & 15;
  const int q0 = (blockIdx.x * 8 + wave) * 16;
  const __bf16* Q = qr + (size_t)bh * TT * DD;
  const __bf16* Kb = kr + (size_t)bh * TT * DD;
  const __bf16* Vb = vt + (size_t)bh * DD * TT;
  __bf16* pb = pbuf + wave * 512;

  // Q A-frags for the two K=32 chunks of D=64 (held in registers all loop)
  const __bf16* qp = Q + (size_t)(q0 + lh) * DD + hi * 16;
  v16bf qf0 = cat16(ld8(qp), ld8(qp + 8));
  v16bf qf1 = cat16(ld8(qp + 32), ld8(qp + 40));

  v8f o0 = {}, o1 = {}, o2 = {}, o3 = {};
  float m[8], l[8];
#pragma unroll
  for (int r = 0; r < 8; ++r) { m[r] = -1e30f; l[r] = 0.0f; }

  for (int k0 = 0; k0 <= q0 + 15; k0 += 32) {
    // scores: 16 queries x 32 keys = two C tiles; K rows feed B-frags directly
    v8f s0 = {}, s1 = {};
    {
      const __bf16* kp0 = Kb + (size_t)(k0 + lh) * DD + hi * 16;
      const __bf16* kp1 = Kb + (size_t)(k0 + 16 + lh) * DD + hi * 16;
      v16bf b00 = cat16(ld8(kp0), ld8(kp0 + 8));
      v16bf b01 = cat16(ld8(kp0 + 32), ld8(kp0 + 40));
      v16bf b10 = cat16(ld8(kp1), ld8(kp1 + 8));
      v16bf b11 = cat16(ld8(kp1 + 32), ld8(kp1 + 40));
      s0 = wmma_bf16(qf0, b00, s0);
      s0 = wmma_bf16(qf1, b01, s0);
      s1 = wmma_bf16(qf0, b10, s1);
      s1 = wmma_bf16(qf1, b11, s1);
    }
    // online softmax over this 32-key block (wave32: 16-lane halves = rows)
#pragma unroll
    for (int r = 0; r < 8; ++r) {
      int qrow = q0 + r + hi * 8;
      float a = s0[r] * 0.125f;            // 1/sqrt(64)
      float bsc = s1[r] * 0.125f;
      if (k0 + lh > qrow) a = -1e30f;      // causal mask
      if (k0 + 16 + lh > qrow) bsc = -1e30f;
      float mx = fmaxf(a, bsc);
      mx = fmaxf(mx, __shfl_xor(mx, 1));
      mx = fmaxf(mx, __shfl_xor(mx, 2));
      mx = fmaxf(mx, __shfl_xor(mx, 4));
      mx = fmaxf(mx, __shfl_xor(mx, 8));
      float mnew = fmaxf(m[r], mx);
      float p0 = __expf(a - mnew), p1 = __expf(bsc - mnew);
      float sum = p0 + p1;
      sum += __shfl_xor(sum, 1);
      sum += __shfl_xor(sum, 2);
      sum += __shfl_xor(sum, 4);
      sum += __shfl_xor(sum, 8);
      float alpha = __expf(m[r] - mnew);
      l[r] = l[r] * alpha + sum;
      m[r] = mnew;
      o0[r] *= alpha; o1[r] *= alpha; o2[r] *= alpha; o3[r] *= alpha;
      // P row-major [16 rows][32 keys] into per-wave LDS (DS ops in-order/wave)
      pb[(r + hi * 8) * 32 + lh] = f2bf(p0);
      pb[(r + hi * 8) * 32 + 16 + lh] = f2bf(p1);
    }
    asm volatile("" ::: "memory");  // keep LDS stores before the frag reload
    // re-read P in A-frag layout (two contiguous 16B LDS loads per lane)
    v16bf pf = cat16(*(const v8bf*)(pb + lh * 32 + hi * 16),
                     *(const v8bf*)(pb + lh * 32 + hi * 16 + 8));
    // V^T rows feed PV B-frags contiguously
    const __bf16* vp = Vb + (size_t)lh * TT + k0 + hi * 16;
    v16bf vf0 = cat16(ld8(vp), ld8(vp + 8));
    v16bf vf1 = cat16(ld8(vp + 16 * TT), ld8(vp + 16 * TT + 8));
    v16bf vf2 = cat16(ld8(vp + 32 * TT), ld8(vp + 32 * TT + 8));
    v16bf vf3 = cat16(ld8(vp + 48 * TT), ld8(vp + 48 * TT + 8));
    o0 = wmma_bf16(pf, vf0, o0);
    o1 = wmma_bf16(pf, vf1, o1);
    o2 = wmma_bf16(pf, vf2, o2);
    o3 = wmma_bf16(pf, vf3, o3);
    asm volatile("" ::: "memory");
  }
  // normalize and scatter to [B,T,H*D] bf16 for the output projection
#pragma unroll
  for (int r = 0; r < 8; ++r) {
    int row = q0 + r + hi * 8;
    float inv = 1.0f / l[r];
    size_t base = ((size_t)(b * TT) + row) * CC + h * DD + lh;
    att[base] = f2bf(o0[r] * inv);
    att[base + 16] = f2bf(o1[r] * inv);
    att[base + 32] = f2bf(o2[r] * inv);
    att[base + 48] = f2bf(o3[r] * inv);
  }
}

// ---------------------------------------------------------------------------
extern "C" void kernel_launch(void* const* d_in, const int* in_sizes, int n_in,
                              void* d_out, int out_size, void* d_ws,
                              size_t ws_size, hipStream_t stream) {
  const float* x  = (const float*)d_in[0];
  const float* Wq = (const float*)d_in[1];
  const float* Wk = (const float*)d_in[2];
  const float* Wv = (const float*)d_in[3];
  const float* Wp = (const float*)d_in[4];
  const float* bp = (const float*)d_in[5];
  float* out = (float*)d_out;

  const int M = MM, K = CC, N = CC;
  __bf16* ws = (__bf16*)d_ws;
  size_t off = 0;
  __bf16* x_bf = ws + off;  off += (size_t)M * K;
  __bf16* WqT  = ws + off;  off += (size_t)N * K;
  __bf16* WkT  = ws + off;  off += (size_t)N * K;
  __bf16* WvT  = ws + off;  off += (size_t)N * K;
  __bf16* WpT  = ws + off;  off += (size_t)N * K;
  __bf16* qlin = ws + off;  off += (size_t)M * N;
  __bf16* klin = ws + off;  off += (size_t)M * N;
  __bf16* vlin = ws + off;  off += (size_t)M * N;
  __bf16* q_r  = ws + off;  off += (size_t)M * N;
  __bf16* k_r  = ws + off;  off += (size_t)M * N;
  __bf16* v_t  = ws + off;  off += (size_t)M * N;
  __bf16* att  = ws + off;  off += (size_t)M * N;   // ~143 MB total

  int nx = M * K;
  cvt_f32_to_bf16<<<(nx + 255) / 256, 256, 0, stream>>>(x, x_bf, nx);

  dim3 tb(32, 32), tg(N / 32, K / 32);
  transpose_cvt<<<tg, tb, 0, stream>>>(Wq, WqT, K, N);
  transpose_cvt<<<tg, tb, 0, stream>>>(Wk, WkT, K, N);
  transpose_cvt<<<tg, tb, 0, stream>>>(Wv, WvT, K, N);
  transpose_cvt<<<tg, tb, 0, stream>>>(Wp, WpT, K, N);

  dim3 gg(N / 64, M / 32 / 8);  // (16, 32), 8 waves/block, 32x64 tile per wave
  wmma_gemm<false><<<gg, 256, 0, stream>>>(x_bf, WqT, nullptr, qlin, M, N, K);
  wmma_gemm<false><<<gg, 256, 0, stream>>>(x_bf, WkT, nullptr, klin, M, N, K);
  wmma_gemm<false><<<gg, 256, 0, stream>>>(x_bf, WvT, nullptr, vlin, M, N, K);

  int nr = BB * TT * HH * (DD / 2);
  rope_pack<<<nr / 256, 256, 0, stream>>>(qlin, klin, vlin, q_r, k_r, v_t);

  dim3 ag(TT / 16 / 8, BB * HH);  // (16, 64)
  flash_attn<<<ag, 256, 0, stream>>>(q_r, k_r, v_t, att);

  wmma_gemm<true><<<gg, 256, 0, stream>>>(att, WpT, bp, out, M, N, K);
}
